// Attention_13898514169968
// MI455X (gfx1250) — matrix operand order repaired and checked by reading, not compiled, not run on hardware
//
#include <hip/hip_runtime.h>

// ---------------- problem constants (from reference) ----------------
#define HEADS   8
#define DH      64
#define DIMSZ   512
#define CSZ     256
#define NBLK    32
#define BSZ     8
#define NTOK    65536          // BSZ * 8192 rows
#define MAXPOS  512

typedef __attribute__((ext_vector_type(16))) _Float16 v16h;
typedef __attribute__((ext_vector_type(8)))  float    v8f;
typedef __attribute__((ext_vector_type(4)))  unsigned uint4v;

struct Frag {
  union { v16h v; _Float16 h[16]; unsigned u[8]; };
};

__device__ __forceinline__ v8f wmma32(const Frag& a, const Frag& b, v8f c) {
  // D = A(16x32 f16) * B(32x16 f16) + C(16x16 f32)
  return __builtin_amdgcn_wmma_f32_16x16x32_f16(false, a.v, false, b.v,
                                                (short)0, c, false, false);
}

// A fragment (16x32, f16 source): p = &src[row_for_lane][k0], row = lane%16 based.
// VGPR r holds K = (r&3)*2 + half*8 + (r>>2)*16  (+0,+1 packed)  [ISA 7.12.2]
__device__ __forceinline__ void loadA16(Frag& f, const _Float16* p, int half) {
#pragma unroll
  for (int r = 0; r < 8; ++r) {
    int k = ((r & 3) << 1) + (half << 3) + ((r >> 2) << 4);
    f.u[r] = *(const unsigned*)(p + k);
  }
}

// A fragment from fp32 source (convert on the fly)
__device__ __forceinline__ void loadA32(Frag& f, const float* p, int half) {
#pragma unroll
  for (int r = 0; r < 8; ++r) {
    int k = ((r & 3) << 1) + (half << 3) + ((r >> 2) << 4);
    f.h[2 * r]     = (_Float16)p[k];
    f.h[2 * r + 1] = (_Float16)p[k + 1];
  }
}

// B fragment (32x16): lane's column fixed (caller folds lane%16 into p);
// K = (r&3)*2 + half*8 + (r>>2)*16 (+0,+1 packed), pairs contiguous along K in source.
__device__ __forceinline__ void loadB16(Frag& f, const _Float16* p, int half) {
  const _Float16* q = p + (half << 3);
#pragma unroll
  for (int r = 0; r < 8; ++r) f.u[r] = *(const unsigned*)(q + ((r & 3) << 1) + ((r >> 2) << 4));
}

// CDNA5 async DMA: copy 16B per lane from global to LDS (ASYNCcnt-tracked).
// INST_OFFSET is applied to BOTH the LDS and the global address (ISA §10,
// GLOBAL_LOAD_ASYNC_TO_LDS pseudocode), so base + offsets copies a run.
__device__ __forceinline__ void async_g2l_b128_0(unsigned lds, const void* g) {
  asm volatile("global_load_async_to_lds_b128 %0, %1, off" ::"v"(lds), "v"(g)
               : "memory");
}
__device__ __forceinline__ void async_g2l_b128_off(unsigned lds, const void* g,
                                                   int imm16) {
  // imm must be a compile-time constant in the unrolled caller
  switch (imm16) {
    case 16:  asm volatile("global_load_async_to_lds_b128 %0, %1, off offset:16"  ::"v"(lds), "v"(g) : "memory"); break;
    case 32:  asm volatile("global_load_async_to_lds_b128 %0, %1, off offset:32"  ::"v"(lds), "v"(g) : "memory"); break;
    case 48:  asm volatile("global_load_async_to_lds_b128 %0, %1, off offset:48"  ::"v"(lds), "v"(g) : "memory"); break;
    case 64:  asm volatile("global_load_async_to_lds_b128 %0, %1, off offset:64"  ::"v"(lds), "v"(g) : "memory"); break;
    case 80:  asm volatile("global_load_async_to_lds_b128 %0, %1, off offset:80"  ::"v"(lds), "v"(g) : "memory"); break;
    case 96:  asm volatile("global_load_async_to_lds_b128 %0, %1, off offset:96"  ::"v"(lds), "v"(g) : "memory"); break;
    case 112: asm volatile("global_load_async_to_lds_b128 %0, %1, off offset:112" ::"v"(lds), "v"(g) : "memory"); break;
    default:  async_g2l_b128_0(lds, g); break;
  }
}
__device__ __forceinline__ void wait_async0() {
  asm volatile("s_wait_asynccnt 0x0" ::: "memory");
}

// ---------------- kernel 0: weight transpose/convert ----------------
__global__ __launch_bounds__(256) void wtrans(const float* __restrict__ W,
                                              _Float16* __restrict__ WT,
                                              int K, int N) {
  int i = blockIdx.x * 256 + threadIdx.x;
  if (i < K * N) {
    int k = i / N, n = i - k * N;
    WT[(size_t)n * K + k] = (_Float16)W[i];       // WT is [N][K]
  }
}

__global__ __launch_bounds__(256) void cvt16(const float* __restrict__ A,
                                             _Float16* __restrict__ B, int n) {
  int i = blockIdx.x * 256 + threadIdx.x;
  if (i < n) B[i] = (_Float16)A[i];
}

// ---------------- kernel 1: x(f32) @ W^T(f16) -> f16 ----------------
__global__ __launch_bounds__(256) void gemm_xw(const float* __restrict__ A,
                                               const _Float16* __restrict__ BT,
                                               _Float16* __restrict__ C,
                                               int K, int N) {
  int tid = threadIdx.x, wave = tid >> 5, lane = tid & 31;
  int lx = lane & 15, half = lane >> 4;
  int m0 = blockIdx.x * 256 + wave * 32;
  int n0 = blockIdx.y * 64;
  v8f z = {};
  v8f acc[2][4];
#pragma unroll
  for (int i = 0; i < 2; ++i)
#pragma unroll
    for (int j = 0; j < 4; ++j) acc[i][j] = z;

  for (int k0 = 0; k0 < K; k0 += 32) {
    if (k0 + 32 < K) {  // GL2 prefetch of next K-chunk (global_prefetch_b8)
      __builtin_prefetch(A + (size_t)(m0 + lx) * K + k0 + 32, 0, 0);
      __builtin_prefetch(BT + (size_t)(n0 + lx) * K + k0 + 32, 0, 0);
    }
    Frag a[2], b[4];
#pragma unroll
    for (int mt = 0; mt < 2; ++mt)
      loadA32(a[mt], A + (size_t)(m0 + mt * 16 + lx) * K + k0, half);
#pragma unroll
    for (int nt = 0; nt < 4; ++nt)
      loadB16(b[nt], BT + (size_t)(n0 + nt * 16 + lx) * K + k0, half);
#pragma unroll
    for (int mt = 0; mt < 2; ++mt)
#pragma unroll
      for (int nt = 0; nt < 4; ++nt)
        acc[mt][nt] = wmma32(a[mt], b[nt], acc[mt][nt]);
  }
#pragma unroll
  for (int mt = 0; mt < 2; ++mt)
#pragma unroll
    for (int nt = 0; nt < 4; ++nt)
#pragma unroll
      for (int r = 0; r < 8; ++r) {
        int row = m0 + mt * 16 + r + half * 8;
        int col = n0 + nt * 16 + lx;
        C[(size_t)row * N + col] = (_Float16)acc[mt][nt][r];
      }
}

// ---------------- kernel 3: ao(f16) @ Wout^T(f16) + b -> f32 ----------------
__global__ __launch_bounds__(256) void gemm_ow(const _Float16* __restrict__ A,
                                               const _Float16* __restrict__ BT,
                                               const float* __restrict__ bias,
                                               float* __restrict__ C,
                                               int K, int N) {
  int tid = threadIdx.x, wave = tid >> 5, lane = tid & 31;
  int lx = lane & 15, half = lane >> 4;
  int m0 = blockIdx.x * 256 + wave * 32;
  int n0 = blockIdx.y * 64;
  v8f z = {};
  v8f acc[2][4];
#pragma unroll
  for (int i = 0; i < 2; ++i)
#pragma unroll
    for (int j = 0; j < 4; ++j) acc[i][j] = z;

  for (int k0 = 0; k0 < K; k0 += 32) {
    if (k0 + 32 < K) {
      __builtin_prefetch(A + (size_t)(m0 + lx) * K + k0 + 32, 0, 0);
      __builtin_prefetch(BT + (size_t)(n0 + lx) * K + k0 + 32, 0, 0);
    }
    Frag a[2], b[4];
#pragma unroll
    for (int mt = 0; mt < 2; ++mt)
      loadA16(a[mt], A + (size_t)(m0 + mt * 16 + lx) * K + k0, half);
#pragma unroll
    for (int nt = 0; nt < 4; ++nt)
      loadB16(b[nt], BT + (size_t)(n0 + nt * 16 + lx) * K + k0, half);
#pragma unroll
    for (int mt = 0; mt < 2; ++mt)
#pragma unroll
      for (int nt = 0; nt < 4; ++nt)
        acc[mt][nt] = wmma32(a[mt], b[nt], acc[mt][nt]);
  }
#pragma unroll
  for (int mt = 0; mt < 2; ++mt)
#pragma unroll
    for (int nt = 0; nt < 4; ++nt) {
      int col = n0 + nt * 16 + lx;
      float bv = bias[col];
#pragma unroll
      for (int r = 0; r < 8; ++r) {
        int row = m0 + mt * 16 + r + half * 8;
        C[(size_t)row * N + col] = acc[mt][nt][r] + bv;
      }
    }
}

// ---------------- kernel 2: local attention, one WG per (b,block,head) ----------------
// LDS: Ks[256][64] f16 | Vt[64][256] f16 | Ps[256][256] f16 | rowmax[256] | invsum[256]
__global__ __launch_bounds__(256) void attn_kernel(const _Float16* __restrict__ qh,
                                                   const _Float16* __restrict__ kvh,
                                                   const _Float16* __restrict__ relh,
                                                   _Float16* __restrict__ ao) {
  extern __shared__ char smem[];
  _Float16* Ks = (_Float16*)smem;            // 16384 halves
  _Float16* Vt = Ks + 256 * 64;              // 16384 halves, [d][key]
  _Float16* Ps = Vt + 64 * 256;              // 65536 halves
  float* rowmax = (float*)(Ps + 256 * 256);  // 256 f32
  float* invsum = rowmax + 256;              // 256 f32

  int tid = threadIdx.x, wave = tid >> 5, lane = tid & 31;
  int lx = lane & 15, half = lane >> 4;
  int bb = blockIdx.x;                 // b*NBLK + blk
  int h  = blockIdx.y;
  int m0w = wave * 32;                 // this wave's 32 query rows (block-local)

  // ---- stage K via async DMA (global -> LDS, no VGPR round trip),
  //      stage V transposed via VGPRs (needs per-element scatter) ----
  {
    int rowg = bb * 256 + tid;
    const _Float16* kvrow = kvh + (size_t)rowg * (2 * DIMSZ) + h * DH;
    unsigned ldsK = (unsigned)(size_t)(Ks + tid * 64);  // low 32 bits = LDS addr
    async_g2l_b128_0(ldsK, kvrow);
#pragma unroll
    for (int c = 1; c < 8; ++c) async_g2l_b128_off(ldsK, kvrow, c * 16);
#pragma unroll
    for (int d8 = 0; d8 < 8; ++d8) {
      union { uint4v u4; _Float16 hh[8]; } t;
      t.u4 = *(const uint4v*)(kvrow + DIMSZ + d8 * 8);
#pragma unroll
      for (int e = 0; e < 8; ++e) Vt[(d8 * 8 + e) * 256 + tid] = t.hh[e];
    }
    wait_async0();
  }
  __syncthreads();

  // ---- Q fragments in registers (2 row-tiles x 2 K-chunks of dh=64) ----
  const _Float16* qbase = qh + (size_t)(bb * 256) * DIMSZ + h * DH;
  Frag Aq[2][2];
#pragma unroll
  for (int mt = 0; mt < 2; ++mt)
#pragma unroll
    for (int kc = 0; kc < 2; ++kc)
      loadA16(Aq[mt][kc], qbase + (size_t)(m0w + mt * 16 + lx) * DIMSZ + kc * 32, half);

  // ---- Pass A: S = (Q K^T + pos) * scale, write raw logits, track row max ----
  float rmax[2][8];
#pragma unroll
  for (int mt = 0; mt < 2; ++mt)
#pragma unroll
    for (int r = 0; r < 8; ++r) rmax[mt][r] = -3.0e38f;

  for (int nt16 = 0; nt16 < 16; ++nt16) {
    int n0 = nt16 * 16;
    Frag Bk[2];
    loadB16(Bk[0], Ks + (n0 + lx) * 64 + 0,  half);
    loadB16(Bk[1], Ks + (n0 + lx) * 64 + 32, half);
#pragma unroll
    for (int mt = 0; mt < 2; ++mt) {
      v8f s = {};
      s = wmma32(Aq[mt][0], Bk[0], s);
      s = wmma32(Aq[mt][1], Bk[1], s);

      // Toeplitz relative-position term: rel row = 512 + (i - j)
      // window base = 497 + m0 - n0, t = (i-m0) - (j-n0) + 15 in [0,30]
      int base = 497 + (m0w + mt * 16) - n0;
      Frag Br0[2], Br1[2];
      loadB16(Br0[0], relh + (size_t)(base + lx) * DH + 0,  half);
      loadB16(Br0[1], relh + (size_t)(base + lx) * DH + 32, half);
      loadB16(Br1[0], relh + (size_t)(base + 16 + lx) * DH + 0,  half);
      loadB16(Br1[1], relh + (size_t)(base + 16 + lx) * DH + 32, half);
      v8f r0 = {}, r1 = {};
      r0 = wmma32(Aq[mt][0], Br0[0], r0);
      r0 = wmma32(Aq[mt][1], Br0[1], r0);
      r1 = wmma32(Aq[mt][0], Br1[0], r1);
      r1 = wmma32(Aq[mt][1], Br1[1], r1);

#pragma unroll
      for (int r = 0; r < 8; ++r) {
        int mi = r + half * 8;           // row within tile (D layout)
        int t  = mi - lx + 15;           // which rel column this element needs
        int srcl = (lane & 16) | (t & 15);
        float g0 = __int_as_float(
            __builtin_amdgcn_ds_bpermute(srcl << 2, __float_as_int(r0[r])));
        float g1 = __int_as_float(
            __builtin_amdgcn_ds_bpermute(srcl << 2, __float_as_int(r1[r])));
        float g  = (t < 16) ? g0 : g1;
        float sv = (s[r] + g) * 0.125f;  // scale = DH^-0.5
        rmax[mt][r] = fmaxf(rmax[mt][r], sv);
        Ps[(m0w + mt * 16 + mi) * 256 + n0 + lx] = (_Float16)sv;
      }
    }
  }
  // reduce row max across the 16 lanes holding each row
#pragma unroll
  for (int mt = 0; mt < 2; ++mt)
#pragma unroll
    for (int r = 0; r < 8; ++r) {
      float m = rmax[mt][r];
      m = fmaxf(m, __shfl_xor(m, 1, 32));
      m = fmaxf(m, __shfl_xor(m, 2, 32));
      m = fmaxf(m, __shfl_xor(m, 4, 32));
      m = fmaxf(m, __shfl_xor(m, 8, 32));
      if (lx == 0) rowmax[m0w + mt * 16 + r + half * 8] = m;
    }
  __syncthreads();

  // ---- Pass B: softmax numerator in place (lane-per-row), vectorized b128 ----
  {
    int row = tid;
    float m = rowmax[row];
    float sum = 0.0f;
    _Float16* prow = Ps + row * 256;
    for (int j = 0; j < 256; j += 8) {
      union { uint4v u4; _Float16 hh[8]; } t;
      t.u4 = *(uint4v*)(prow + j);
#pragma unroll
      for (int e = 0; e < 8; ++e) {
        float ev = __expf((float)t.hh[e] - m);
        sum += ev;
        t.hh[e] = (_Float16)ev;
      }
      *(uint4v*)(prow + j) = t.u4;
    }
    invsum[row] = 1.0f / sum;
  }
  __syncthreads();

  // ---- Pass C: O = P @ V, normalize, store f16 to attn_out ----
  v8f z = {};
  v8f o[2][4];
#pragma unroll
  for (int i = 0; i < 2; ++i)
#pragma unroll
    for (int j = 0; j < 4; ++j) o[i][j] = z;

  for (int j0 = 0; j0 < 256; j0 += 32) {
    Frag Bv[4];
#pragma unroll
    for (int dt = 0; dt < 4; ++dt)
      loadB16(Bv[dt], Vt + (dt * 16 + lx) * 256 + j0, half);
#pragma unroll
    for (int mt = 0; mt < 2; ++mt) {
      Frag Ap;
      loadA16(Ap, Ps + (m0w + mt * 16 + lx) * 256 + j0, half);
#pragma unroll
      for (int dt = 0; dt < 4; ++dt)
        o[mt][dt] = wmma32(Ap, Bv[dt], o[mt][dt]);
    }
  }

  _Float16* aobase = ao + (size_t)(bb * 256) * DIMSZ + h * DH;
#pragma unroll
  for (int mt = 0; mt < 2; ++mt)
#pragma unroll
    for (int r = 0; r < 8; ++r) {
      int row = m0w + mt * 16 + r + half * 8;
      float inv = invsum[row];
#pragma unroll
      for (int dt = 0; dt < 4; ++dt)
        aobase[(size_t)row * DIMSZ + dt * 16 + lx] = (_Float16)(o[mt][dt][r] * inv);
    }
}

// ---------------- host orchestration ----------------
extern "C" void kernel_launch(void* const* d_in, const int* in_sizes, int n_in,
                              void* d_out, int out_size, void* d_ws, size_t ws_size,
                              hipStream_t stream) {
  (void)in_sizes; (void)n_in; (void)out_size; (void)ws_size;
  const float* x    = (const float*)d_in[0];   // [8,8192,512]
  const float* Wq   = (const float*)d_in[1];   // [512,512]
  const float* Wkv  = (const float*)d_in[2];   // [512,1024]
  const float* Wout = (const float*)d_in[3];   // [512,512]
  const float* bout = (const float*)d_in[4];   // [512]
  const float* rel  = (const float*)d_in[5];   // [1025,64]
  float* out = (float*)d_out;                  // [8,8192,512] f32

  // workspace carve-up (all f16)
  char* p = (char*)d_ws;
  _Float16* WqT  = (_Float16*)p; p += (size_t)512 * 512 * 2;
  _Float16* WkvT = (_Float16*)p; p += (size_t)512 * 1024 * 2;
  _Float16* WoT  = (_Float16*)p; p += (size_t)512 * 512 * 2;
  _Float16* relh = (_Float16*)p; p += (size_t)1025 * 64 * 2 + 128;
  _Float16* qhp  = (_Float16*)p; p += (size_t)NTOK * 512 * 2;
  _Float16* kvhp = (_Float16*)p; p += (size_t)NTOK * 1024 * 2;
  _Float16* aop  = (_Float16*)p;

  // 0) convert/transpose weights and rel table
  wtrans<<<dim3((512 * 512 + 255) / 256), dim3(256), 0, stream>>>(Wq, WqT, 512, 512);
  wtrans<<<dim3((512 * 1024 + 255) / 256), dim3(256), 0, stream>>>(Wkv, WkvT, 512, 1024);
  wtrans<<<dim3((512 * 512 + 255) / 256), dim3(256), 0, stream>>>(Wout, WoT, 512, 512);
  cvt16<<<dim3((1025 * 64 + 255) / 256), dim3(256), 0, stream>>>(rel, relh, 1025 * 64);

  // 1) projections (WMMA f16)
  gemm_xw<<<dim3(NTOK / 256, 512 / 64), dim3(256), 0, stream>>>(x, WqT, qhp, 512, 512);
  gemm_xw<<<dim3(NTOK / 256, 1024 / 64), dim3(256), 0, stream>>>(x, WkvT, kvhp, 512, 1024);

  // 2) attention: one WG per (b, block, head); 198.7 KB dynamic LDS
  size_t smem = (size_t)(256 * 64 + 64 * 256 + 256 * 256) * 2 + 512 * 4;
  (void)hipFuncSetAttribute((const void*)attn_kernel,
                            hipFuncAttributeMaxDynamicSharedMemorySize, (int)smem);
  attn_kernel<<<dim3(BSZ * NBLK, HEADS), dim3(256), smem, stream>>>(qhp, kvhp, relh, aop);

  // 3) output projection + bias (WMMA f16, f32 out)
  gemm_ow<<<dim3(NTOK / 256, 512 / 64), dim3(256), 0, stream>>>(aop, WoT, bout, out, 512, 512);
}
